// MoE_14173392077387
// MI455X (gfx1250) — compile-verified, matrix operand order checked by reading, not run on hardware
//
#include <hip/hip_runtime.h>
#include <math.h>

// ---------------- problem constants (from reference) ----------------
constexpr int NTOK = 16384;          // B*T
constexpr int DMODEL = 1024;
constexpr int NEXP = 8;
constexpr int DFF = 4096;
constexpr int TOPK = 2;
constexpr int CAP = 8192;            // 2*N*K/E
constexpr int MCHUNK = 2048;         // rows of capacity processed per GEMM pass
constexpr int NCHUNK = CAP / MCHUNK; // 4

typedef unsigned short ushort_t;
typedef unsigned int uint_t;
typedef __bf16 bf16x16 __attribute__((ext_vector_type(16)));
typedef float v8f __attribute__((ext_vector_type(8)));

// ---------------- helpers ----------------
__device__ __forceinline__ ushort_t f2bf(float f) {
    uint_t u = __float_as_uint(f);
    u += 0x7FFFu + ((u >> 16) & 1u);     // round-to-nearest-even
    return (ushort_t)(u >> 16);
}
__device__ __forceinline__ float gelu_tanh(float v) {
    float c = 0.7978845608028654f;
    return 0.5f * v * (1.0f + tanhf(c * (v + 0.044715f * v * v * v)));
}
__device__ __forceinline__ float softplus(float z) {
    return (z > 20.0f) ? z : log1pf(__expf(z));
}

// ---------------- workspace layout (bytes) ----------------
constexpr size_t OFF_W1C = 0;                                   // bf16 [E][d][dff]
constexpr size_t SZ_W1C  = (size_t)NEXP * DMODEL * DFF * 2;
constexpr size_t OFF_W2C = OFF_W1C + SZ_W1C;                    // bf16 [E][dff][d]
constexpr size_t SZ_W2C  = (size_t)NEXP * DFF * DMODEL * 2;
constexpr size_t OFF_BUF = OFF_W2C + SZ_W2C;                    // bf16 [E][CAP][d]
constexpr size_t SZ_BUF  = (size_t)NEXP * CAP * DMODEL * 2;
constexpr size_t OFF_HCH = OFF_BUF + SZ_BUF;                    // bf16 [E][MCHUNK][dff]
constexpr size_t SZ_HCH  = (size_t)NEXP * MCHUNK * DFF * 2;
constexpr size_t OFF_TOK = OFF_HCH + SZ_HCH;                    // int  [E][CAP]
constexpr size_t SZ_TOK  = (size_t)NEXP * CAP * 4;
constexpr size_t OFF_WPR = OFF_TOK + SZ_TOK;                    // f32  [E][CAP]
constexpr size_t SZ_WPR  = (size_t)NEXP * CAP * 4;
constexpr size_t OFF_PMT = OFF_WPR + SZ_WPR;                    // f32  [N][E]
constexpr size_t SZ_PMT  = (size_t)NTOK * NEXP * 4;
constexpr size_t OFF_CNT = OFF_PMT + SZ_PMT;                    // int  [E]

// ---------------- zero / convert ----------------
__global__ void zero_kernel(float* __restrict__ out, long long n, int* __restrict__ cnt) {
    long long i = (long long)blockIdx.x * blockDim.x + threadIdx.x;
    long long stride = (long long)gridDim.x * blockDim.x;
    for (; i < n; i += stride) out[i] = 0.0f;
    if (blockIdx.x == 0 && threadIdx.x < NEXP) cnt[threadIdx.x] = 0;
}

__global__ void cvt_kernel(const float* __restrict__ src, ushort_t* __restrict__ dst, long long n) {
    long long i = (long long)blockIdx.x * blockDim.x + threadIdx.x;
    long long stride = (long long)gridDim.x * blockDim.x;
    for (; i < n; i += stride) dst[i] = f2bf(src[i]);
}

// ---------------- gating / routing / dispatch (one wave32 per token) ----------------
__global__ __launch_bounds__(256)
void gate_kernel(const float* __restrict__ x, const float* __restrict__ noise,
                 const float* __restrict__ gw, const float* __restrict__ nw,
                 ushort_t* __restrict__ buf, int* __restrict__ slotTok,
                 float* __restrict__ slotW, int* __restrict__ counters,
                 float* __restrict__ pmat) {
    int lane = threadIdx.x & 31;
    int wave = threadIdx.x >> 5;
    int t = blockIdx.x * 8 + wave;
    if (t >= NTOK) return;

    float g[NEXP], nn[NEXP];
#pragma unroll
    for (int e = 0; e < NEXP; ++e) { g[e] = 0.f; nn[e] = 0.f; }

    const float* xr = x + (size_t)t * DMODEL;
    for (int i = lane; i < DMODEL; i += 32) {
        float xv = xr[i];
        const float* gr = gw + (size_t)i * NEXP;
        const float* nr = nw + (size_t)i * NEXP;
#pragma unroll
        for (int e = 0; e < NEXP; ++e) { g[e] = fmaf(xv, gr[e], g[e]); nn[e] = fmaf(xv, nr[e], nn[e]); }
    }
#pragma unroll
    for (int off = 16; off > 0; off >>= 1) {
#pragma unroll
        for (int e = 0; e < NEXP; ++e) {
            g[e]  += __shfl_xor(g[e],  off, 32);
            nn[e] += __shfl_xor(nn[e], off, 32);
        }
    }

    // noisy logits (all lanes hold identical values)
    float lg[NEXP], sd[NEXP];
#pragma unroll
    for (int e = 0; e < NEXP; ++e) {
        sd[e] = softplus(nn[e]);
        lg[e] = g[e] + noise[(size_t)t * NEXP + e] * sd[e];
    }

    // top-3 (values + top-2 indices)
    float v1 = -1e30f, v2 = -1e30f, v3 = -1e30f; int i1 = 0, i2 = 0;
#pragma unroll
    for (int e = 0; e < NEXP; ++e) {
        float v = lg[e];
        if (v > v1)      { v3 = v2; v2 = v1; i2 = i1; v1 = v; i1 = e; }
        else if (v > v2) { v3 = v2; v2 = v; i2 = e; }
        else if (v > v3) { v3 = v; }
    }

    // load probability for expert == lane (deterministic per-token store)
    if (lane < NEXP) {
        int e = lane;
        float kth = (e == i1 || e == i2) ? v3 : v2;
        float z = (lg[e] - kth) / sd[e];
        float P = 0.5f * (1.0f + erff(z * 0.70710678118654752f));
        pmat[(size_t)t * NEXP + e] = P;
    }

    // softmax over top-2
    float e2 = __expf(v2 - v1);
    float inv = 1.0f / (1.0f + e2);
    float w0 = inv, w1 = e2 * inv;

#pragma unroll
    for (int k = 0; k < TOPK; ++k) {
        int e = k ? i2 : i1;
        float w = k ? w1 : w0;
        int pos = 0;
        if (lane == 0) pos = atomicAdd(&counters[e], 1);
        pos = __shfl(pos, 0, 32);
        if (pos < CAP) {
            if (lane == 0) {
                slotTok[(size_t)e * CAP + pos] = t;
                slotW[(size_t)e * CAP + pos]   = w;
            }
            ushort_t* dst = buf + ((size_t)e * CAP + pos) * DMODEL;
            for (int i = lane; i < DMODEL; i += 32) dst[i] = f2bf(xr[i]);
        }
    }
}

// ---------------- load-balance loss (deterministic serial reduction) ----------------
__global__ void loss_kernel(const float* __restrict__ pmat, float* __restrict__ lossOut) {
    __shared__ float s[NEXP];
    int tid = threadIdx.x;
    if (tid < NEXP) {
        double acc = 0.0;
        for (int t = 0; t < NTOK; ++t) acc += (double)pmat[(size_t)t * NEXP + tid];
        s[tid] = (float)acc;
    }
    __syncthreads();
    if (tid == 0) {
        double m = 0.0;
        for (int e = 0; e < NEXP; ++e) m += s[e];
        m /= NEXP;
        double var = 0.0;
        for (int e = 0; e < NEXP; ++e) { double d = s[e] - m; var += d * d; }
        var /= (NEXP - 1);                       // unbiased, like torch .std()
        double cv2 = var / (m * m);
        *lossOut = (float)(0.01 * cv2);          // W_LOAD * cv^2
    }
}

// ---------------- WMMA grouped GEMM (double-buffered, async A staging) ----------------
// Block: 256 thr = 8 wave32, tile 128(M) x 256(N), K-step 32, wave tile 64x64.
// A tile: global -> LDS via GLOBAL_LOAD_ASYNC_TO_LDS_B128 (ASYNCcnt), ping-pong buffers.
// B tile: global -> regs -> transposed ds_store, ping-pong buffers.
// EPI 0: h = gelu(A*B + b1)  -> bf16 hOut
// EPI 1: scatter (A*B + b2)*slotW into out rows via global f32 atomics
union Frag { uint4 q[2]; bf16x16 v; };
union Acc  { v8f v; float f[8]; };
struct BReg { uint4 v[4]; };

__device__ __forceinline__ void stageA_async(const ushort_t* Ae, int rowBase, int K, int k0,
                                             ushort_t* bufA, int tid) {
#pragma unroll
    for (int r2 = 0; r2 < 2; ++r2) {
        int q = tid + r2 * 256;
        int row = q >> 2, seg = q & 3;
        const ushort_t* gp = Ae + (size_t)(rowBase + row) * K + k0 + seg * 8;
        unsigned lds = (unsigned)(size_t)(bufA + row * 40 + seg * 8);
        asm volatile("global_load_async_to_lds_b128 %0, %1, off"
                     :: "v"(lds), "v"(gp) : "memory");
    }
}
__device__ __forceinline__ void loadB_regs(const ushort_t* Be, int nBase, int N, int k0,
                                           int tid, BReg& r) {
#pragma unroll
    for (int c2 = 0; c2 < 4; ++c2) {
        int q = tid + c2 * 256;
        int kk = q >> 5, ng = q & 31;
        r.v[c2] = *(const uint4*)(Be + (size_t)(k0 + kk) * N + nBase + ng * 8);
    }
}
__device__ __forceinline__ void storeB_lds(ushort_t* bufB, int tid, const BReg& r) {
#pragma unroll
    for (int c2 = 0; c2 < 4; ++c2) {
        int q = tid + c2 * 256;
        int kk = q >> 5, ng = q & 31;
        union { uint4 u; ushort_t s[8]; } tv;
        tv.u = r.v[c2];
#pragma unroll
        for (int j = 0; j < 8; ++j) bufB[(ng * 8 + j) * 40 + kk] = tv.s[j];
    }
}
__device__ __forceinline__ void wait_async_all() {
    asm volatile("s_wait_asynccnt 0x0" ::: "memory");
}

template <int EPI>
__global__ __launch_bounds__(256)
void gemm_wmma(const ushort_t* __restrict__ A, long long aStrideE,
               const ushort_t* __restrict__ B, long long bStrideE,
               const float* __restrict__ bias, int biasStrideE,
               int N, int K,
               ushort_t* __restrict__ hOut, long long hStrideE,
               float* __restrict__ out, const int* __restrict__ slotTok,
               const float* __restrict__ slotW, const int* __restrict__ counters,
               int slotBase) {
    __shared__ alignas(16) ushort_t lsA[2][128 * 40];   // rows padded to 80B (16B aligned)
    __shared__ alignas(16) ushort_t lsB[2][256 * 40];   // B staged transposed: [n][k]

    const int tid   = threadIdx.x;
    const int lane  = tid & 31;
    const int wave  = tid >> 5;
    const int waveM = wave >> 2;         // 0..1
    const int waveN = wave & 3;          // 0..3
    const int half  = lane >> 4;
    const int l15   = lane & 15;

    const int e       = blockIdx.z;
    const int nBase   = blockIdx.x * 256;
    const int rowBase = blockIdx.y * 128;

    const ushort_t* Ae = A + (size_t)e * aStrideE;
    const ushort_t* Be = B + (size_t)e * bStrideE;
    const float*    be = bias + (size_t)e * biasStrideE;

    v8f acc[4][4];
#pragma unroll
    for (int i = 0; i < 4; ++i)
#pragma unroll
        for (int j = 0; j < 4; ++j)
#pragma unroll
            for (int v = 0; v < 8; ++v) acc[i][j][v] = 0.0f;

    int cnt = 0;
    if (EPI == 1) cnt = min(counters[e], CAP);

    const int S = K / 32;
    BReg br;

    // prologue: stage tile 0 into buffer 0
    stageA_async(Ae, rowBase, K, 0, &lsA[0][0], tid);
    loadB_regs(Be, nBase, N, 0, tid, br);
    storeB_lds(&lsB[0][0], tid, br);
    wait_async_all();
    __syncthreads();

    for (int s = 0; s < S; ++s) {
        const int cur = s & 1;
        const int nxt = cur ^ 1;
        if (s + 1 < S) {
            stageA_async(Ae, rowBase, K, (s + 1) * 32, &lsA[nxt][0], tid);
            loadB_regs(Be, nBase, N, (s + 1) * 32, tid, br);
        }

        // compute on current buffers
        Frag fa[4], fb[4];
#pragma unroll
        for (int mi = 0; mi < 4; ++mi) {
            int r = waveM * 64 + mi * 16 + l15;
            int b = r * 40 + half * 8;                     // bytes: r*80 + half*16
            fa[mi].q[0] = *(const uint4*)(&lsA[cur][b]);
            fa[mi].q[1] = *(const uint4*)(&lsA[cur][b + 16]);  // K upper half (+32B)
        }
#pragma unroll
        for (int nj = 0; nj < 4; ++nj) {
            int r = waveN * 64 + nj * 16 + l15;
            int b = r * 40 + half * 8;
            fb[nj].q[0] = *(const uint4*)(&lsB[cur][b]);
            fb[nj].q[1] = *(const uint4*)(&lsB[cur][b + 16]);
        }
#pragma unroll
        for (int mi = 0; mi < 4; ++mi)
#pragma unroll
            for (int nj = 0; nj < 4; ++nj)
                acc[mi][nj] = __builtin_amdgcn_wmma_f32_16x16x32_bf16(
                    false, fa[mi].v, false, fb[nj].v, (short)0, acc[mi][nj], false, false);

        if (s + 1 < S) {
            storeB_lds(&lsB[nxt][0], tid, br);
            wait_async_all();       // this wave's async A copies for step s+1 landed
            __syncthreads();        // all waves' A async + B ds stores visible
        }
    }

    // epilogue: C layout -> row = v + 8*half, col = l15
#pragma unroll
    for (int mi = 0; mi < 4; ++mi) {
#pragma unroll
        for (int nj = 0; nj < 4; ++nj) {
            Acc r; r.v = acc[mi][nj];
#pragma unroll
            for (int v = 0; v < 8; ++v) {
                int row = rowBase + waveM * 64 + mi * 16 + v + half * 8;  // chunk-local
                int col = nBase + waveN * 64 + nj * 16 + l15;
                if (EPI == 0) {
                    float val = gelu_tanh(r.f[v] + be[col]);
                    hOut[(size_t)e * hStrideE + (size_t)row * N + col] = f2bf(val);
                } else {
                    int slot = slotBase + row;
                    if (slot < cnt) {
                        int   t = slotTok[(size_t)e * CAP + slot];
                        float w = slotW[(size_t)e * CAP + slot];
                        atomicAdd(out + (size_t)t * DMODEL + col, (r.f[v] + be[col]) * w);
                    }
                }
            }
        }
    }
}

// ---------------- launch ----------------
extern "C" void kernel_launch(void* const* d_in, const int* in_sizes, int n_in,
                              void* d_out, int out_size, void* d_ws, size_t ws_size,
                              hipStream_t stream) {
    const float* x      = (const float*)d_in[0];
    const float* noise  = (const float*)d_in[1];
    const float* gate_w = (const float*)d_in[2];
    const float* nois_w = (const float*)d_in[3];
    const float* w1     = (const float*)d_in[4];
    const float* b1     = (const float*)d_in[5];
    const float* w2     = (const float*)d_in[6];
    const float* b2     = (const float*)d_in[7];
    float* out = (float*)d_out;

    char* ws = (char*)d_ws;
    ushort_t* w1c = (ushort_t*)(ws + OFF_W1C);
    ushort_t* w2c = (ushort_t*)(ws + OFF_W2C);
    ushort_t* buf = (ushort_t*)(ws + OFF_BUF);
    ushort_t* hch = (ushort_t*)(ws + OFF_HCH);
    int*    slotTok = (int*)(ws + OFF_TOK);
    float*  slotW   = (float*)(ws + OFF_WPR);
    float*  pmat    = (float*)(ws + OFF_PMT);
    int*    cnt     = (int*)(ws + OFF_CNT);

    const long long outElems = (long long)NTOK * DMODEL;   // token outputs (loss written later)

    zero_kernel<<<4096, 256, 0, stream>>>(out, outElems, cnt);
    cvt_kernel<<<8192, 256, 0, stream>>>(w1, w1c, (long long)NEXP * DMODEL * DFF);
    cvt_kernel<<<8192, 256, 0, stream>>>(w2, w2c, (long long)NEXP * DFF * DMODEL);

    gate_kernel<<<NTOK / 8, 256, 0, stream>>>(x, noise, gate_w, nois_w,
                                              buf, slotTok, slotW, cnt, pmat);

    loss_kernel<<<1, 32, 0, stream>>>(pmat, out + (out_size - 1));

    for (int cc = 0; cc < NCHUNK; ++cc) {
        const ushort_t* Abuf = buf + (size_t)cc * MCHUNK * DMODEL;
        // GEMM1: h = gelu(buf * w1 + b1), per expert, chunk of MCHUNK rows
        gemm_wmma<0><<<dim3(DFF / 256, MCHUNK / 128, NEXP), 256, 0, stream>>>(
            Abuf, (long long)CAP * DMODEL,
            w1c, (long long)DMODEL * DFF,
            b1, DFF,
            DFF, DMODEL,
            hch, (long long)MCHUNK * DFF,
            nullptr, nullptr, nullptr, nullptr, 0);
        // GEMM2: scatter (h * w2 + b2) * slotW into token rows
        gemm_wmma<1><<<dim3(DMODEL / 256, MCHUNK / 128, NEXP), 256, 0, stream>>>(
            hch, (long long)MCHUNK * DFF,
            w2c, (long long)DFF * DMODEL,
            b2, DMODEL,
            DMODEL, DFF,
            nullptr, 0,
            out, slotTok, slotW, cnt, cc * MCHUNK);
    }
}
